// FusionModel_88158498718091
// MI455X (gfx1250) — matrix operand order repaired and checked by reading, not compiled, not run on hardware
//
#include <hip/hip_runtime.h>
#include <hip/hip_bf16.h>
#include <stdint.h>

// ---------------------------------------------------------------------------
// Problem constants (match reference)
// ---------------------------------------------------------------------------
#define BATCH 256
#define DIN   63
#define D1    64          // DIN + ones column
#define H1    256
#define H2    128
#define NSLAB 4096        // D1*D1 (i,j) pairs; each slab contributes K=64
#define NEG_SLOPE 0.1f
#define BN_EPS    1e-5f

// Tiling
#define NTILE        64   // H1 columns per workgroup
#define SLABS_PER_WG 16   // K per WG = 16*64 = 1024
#define NSPLIT       (NSLAB / SLABS_PER_WG)   // 256 K-splits
#define WSTRIDE      65   // padded fp32 dword stride: 8*65 % 64 == 8 -> lane
                          // halves of a B-fragment read (k 8 apart) start 8 banks apart

typedef __attribute__((ext_vector_type(16))) __bf16    v16bf;
typedef __attribute__((ext_vector_type(8)))  float     v8f;

// ---------------------------------------------------------------------------
// CDNA5 async global->LDS copy (ASYNCcnt-tracked, overlaps with WMMA compute)
// ---------------------------------------------------------------------------
__device__ __forceinline__ void async_g2l_b32(const float* gsrc, const float* lds_dst) {
    // Low 32 bits of a generic shared pointer == wave-relative LDS byte offset.
    const uint32_t l = (uint32_t)(uintptr_t)lds_dst;
    asm volatile("global_load_async_to_lds_b32 %0, %1, off"
                 :: "v"(l), "v"(gsrc) : "memory");
}
__device__ __forceinline__ void wait_async_le(int prev_batch_outstanding) {
    if (prev_batch_outstanding)
        asm volatile("s_wait_asynccnt 0x10" ::: "memory");  // <=16: prev slab done
    else
        asm volatile("s_wait_asynccnt 0x0" ::: "memory");   // last slab: drain
}

// ---------------------------------------------------------------------------
// Kernel 0: zero the split-K accumulator (d_ws is poisoned by the harness)
// ---------------------------------------------------------------------------
__global__ __launch_bounds__(256) void k0_zero(float* __restrict__ acc) {
    acc[blockIdx.x * 256 + threadIdx.x] = 0.0f;
}

// ---------------------------------------------------------------------------
// Kernel 1: implicit-Kronecker GEMM   acc[b,n] += sum_f x[b,f] * W1[f,n]
//   grid.x = NSPLIT * (H1/NTILE) = 1024, block = 256 (8 waves)
//   Per slab (i,j):  P[b,n] = sum_k v3e[b,k] * W1[slab*64+k, n]   (bf16 WMMA)
//                    cacc[b,n] += (v1e[b,i]*v2e[b,j]) * P[b,n]    (fp32 diag scale)
// ---------------------------------------------------------------------------
__global__ __launch_bounds__(256)
__attribute__((amdgpu_waves_per_eu(4)))     // cap at 256 VGPRs: no vgpr-msb SALU,
void k1_kron_gemm(                          // 4 waves/SIMD for HBM latency hiding
    const float* __restrict__ v1, const float* __restrict__ v2,
    const float* __restrict__ v3, const float* __restrict__ W1,
    float* __restrict__ acc)
{
    __shared__ float s_w[2][64 * WSTRIDE];   // double-buffered raw fp32 W slab, 32.5 KB
    __shared__ float s_s12[2][BATCH];        // double-buffered v1[:,i]*v2[:,j], 2 KB

    const int tid    = threadIdx.x;
    const int ntile  = blockIdx.x & 3;       // which 64-column tile of H1
    const int ksplit = blockIdx.x >> 2;      // which group of 16 slabs
    const int n_base = ntile * NTILE;
    const int slab0  = ksplit * SLABS_PER_WG;

    const int wave = tid >> 5;
    const int lane = tid & 31;
    const int kh   = lane >> 4;              // lane half
    const int lm   = lane & 15;
    const int m0   = wave * 32;              // wave owns batch rows m0..m0+31

    // ---- Resident A fragments: pure v3 (+ones col), slab-invariant ----
    // ISA 16-bit A 16x32 layout: VGPR v holds k = 16*(v>>2) + 8*kh + 2*(v&3)
    v16bf afr[2][2];
    #pragma unroll
    for (int t = 0; t < 2; ++t) {
        const int m = m0 + t * 16 + lm;
        #pragma unroll
        for (int ks = 0; ks < 2; ++ks)
            #pragma unroll
            for (int v = 0; v < 8; ++v) {
                const int k = ks * 32 + 16 * (v >> 2) + 8 * kh + 2 * (v & 3);
                const float f0 = (k     < DIN) ? v3[m * DIN + k]     : 1.0f;
                const float f1 = (k + 1 < DIN) ? v3[m * DIN + k + 1] : 1.0f;
                afr[t][ks][2 * v]     = (__bf16)f0;
                afr[t][ks][2 * v + 1] = (__bf16)f1;
            }
    }

    v8f cacc[2][4];
    const v8f vzero = {0.f,0.f,0.f,0.f,0.f,0.f,0.f,0.f};
    #pragma unroll
    for (int t = 0; t < 2; ++t)
        #pragma unroll
        for (int u = 0; u < 4; ++u) cacc[t][u] = vzero;

    // ---- Prologue: start async copy of slab 0, compute its s12 ----
    {
        const float* wsrc = W1 + (size_t)slab0 * 64 * H1 + n_base;
        #pragma unroll
        for (int c = 0; c < 16; ++c) {
            const int e = c * 256 + tid, k = e >> 6, n = e & 63;
            async_g2l_b32(wsrc + (size_t)k * H1 + n, &s_w[0][k * WSTRIDE + n]);
        }
        const int i = slab0 >> 6, j = slab0 & 63;
        const float a = (i < DIN) ? v1[tid * DIN + i] : 1.0f;
        const float c = (j < DIN) ? v2[tid * DIN + j] : 1.0f;
        s_s12[0][tid] = a * c;
    }

    for (int s = 0; s < SLABS_PER_WG; ++s) {
        const int cur = s & 1;
        __syncthreads();   // all waves done reading s_w[1-cur] from iteration s-1

        if (s + 1 < SLABS_PER_WG) {
            // Stream next slab into the other buffer while this one computes.
            const int slab1 = slab0 + s + 1;
            const float* wsrc = W1 + (size_t)slab1 * 64 * H1 + n_base;
            #pragma unroll
            for (int c = 0; c < 16; ++c) {
                const int e = c * 256 + tid, k = e >> 6, n = e & 63;
                async_g2l_b32(wsrc + (size_t)k * H1 + n,
                              &s_w[1 - cur][k * WSTRIDE + n]);
            }
            const int i = slab1 >> 6, j = slab1 & 63;
            const float a = (i < DIN) ? v1[tid * DIN + i] : 1.0f;
            const float c = (j < DIN) ? v2[tid * DIN + j] : 1.0f;
            s_s12[1 - cur][tid] = a * c;
            // L2 prefetch two slabs ahead.
            if (s + 2 < SLABS_PER_WG && tid < 64)
                __builtin_prefetch(W1 + (size_t)(slab1 + 1) * 64 * H1 + n_base
                                   + (size_t)tid * H1, 0, 1);
            wait_async_le(16);   // slab s's 16 copies (in-order) have landed
        } else {
            wait_async_le(0);    // drain
        }
        __syncthreads();         // every wave's copies for `cur` are in LDS

        // ---- Per-slab compute ----
        const float* wb = s_w[cur];
        float sr[2][8];
        #pragma unroll
        for (int t = 0; t < 2; ++t)
            #pragma unroll
            for (int r = 0; r < 8; ++r)
                sr[t][r] = s_s12[cur][m0 + t * 16 + 8 * kh + r];

        // N-subtile-major order: P lives for only one subtile (16 VGPRs),
        // immediately folded into cacc with the fp32 diagonal scale.
        #pragma unroll
        for (int u = 0; u < 4; ++u) {
            v8f P[2];
            P[0] = vzero;
            P[1] = vzero;
            #pragma unroll
            for (int ks = 0; ks < 2; ++ks) {
                // ISA 16-bit B 32x16 layout: n = lane%16, VGPR v: k = 16*(v>>2) + 8*kh + 2*(v&3)
                const int n = u * 16 + lm;
                v16bf bfr;
                #pragma unroll
                for (int v = 0; v < 8; ++v) {
                    const int k = ks * 32 + 16 * (v >> 2) + 8 * kh + 2 * (v & 3);
                    bfr[2 * v]     = (__bf16)wb[k * WSTRIDE + n];
                    bfr[2 * v + 1] = (__bf16)wb[(k + 1) * WSTRIDE + n];
                }
                #pragma unroll
                for (int t = 0; t < 2; ++t)
                    P[t] = __builtin_amdgcn_wmma_f32_16x16x32_bf16(
                        false, afr[t][ks], false, bfr,
                        (short)0, P[t], false, false);
            }
            #pragma unroll
            for (int t = 0; t < 2; ++t)
                #pragma unroll
                for (int r = 0; r < 8; ++r)
                    cacc[t][u][r] += sr[t][r] * P[t][r];
        }
    }

    // Split-K reduction: fp32 global atomics into acc[BATCH][H1].
    // C/D layout: VGPR r -> m = r + 8*kh, n = lane%16.
    #pragma unroll
    for (int t = 0; t < 2; ++t) {
        const int mb = m0 + t * 16 + 8 * kh;
        #pragma unroll
        for (int u = 0; u < 4; ++u) {
            const int n = n_base + u * 16 + lm;
            #pragma unroll
            for (int r = 0; r < 8; ++r)
                atomicAdd(&acc[(size_t)(mb + r) * H1 + n], cacc[t][u][r]);
        }
    }
}

// ---------------------------------------------------------------------------
// Kernel 2: fused tail — bias+leaky+BN1, GEMM2 (256x128), leaky+BN2, classifier
// ---------------------------------------------------------------------------
__global__ __launch_bounds__(128) void k2_tail(
    const float* __restrict__ acc,
    const float* __restrict__ b1, const float* __restrict__ g1,
    const float* __restrict__ be1, const float* __restrict__ rm1,
    const float* __restrict__ rv1,
    const float* __restrict__ W2, const float* __restrict__ b2,
    const float* __restrict__ g2, const float* __restrict__ be2,
    const float* __restrict__ rm2, const float* __restrict__ rv2,
    const float* __restrict__ Wc, const float* __restrict__ bc,
    float* __restrict__ out)
{
    __shared__ float a1[H1];
    __shared__ float red[H2];
    const int b = blockIdx.x, t = threadIdx.x;

    for (int m = t; m < H1; m += H2) {
        float x = acc[(size_t)b * H1 + m] + b1[m];
        x = (x >= 0.f) ? x : NEG_SLOPE * x;
        x = (x - rm1[m]) * rsqrtf(rv1[m] + BN_EPS) * g1[m] + be1[m];
        a1[m] = x;
    }
    __syncthreads();

    float z = b2[t];
    #pragma unroll 4
    for (int m = 0; m < H1; ++m) z += a1[m] * W2[m * H2 + t];
    z = (z >= 0.f) ? z : NEG_SLOPE * z;
    z = (z - rm2[t]) * rsqrtf(rv2[t] + BN_EPS) * g2[t] + be2[t];

    red[t] = z * Wc[t];
    __syncthreads();
    for (int off = H2 / 2; off > 0; off >>= 1) {
        if (t < off) red[t] += red[t + off];
        __syncthreads();
    }
    if (t == 0) out[b] = red[0] + bc[0];
}

// ---------------------------------------------------------------------------
extern "C" void kernel_launch(void* const* d_in, const int* in_sizes, int n_in,
                              void* d_out, int out_size, void* d_ws, size_t ws_size,
                              hipStream_t stream) {
    const float* v1   = (const float*)d_in[0];
    const float* v2   = (const float*)d_in[1];
    const float* v3   = (const float*)d_in[2];
    const float* W1   = (const float*)d_in[3];
    const float* b1   = (const float*)d_in[4];
    const float* g1   = (const float*)d_in[5];
    const float* be1  = (const float*)d_in[6];
    const float* rm1  = (const float*)d_in[7];
    const float* rv1  = (const float*)d_in[8];
    const float* W2   = (const float*)d_in[9];
    const float* b2   = (const float*)d_in[10];
    const float* g2   = (const float*)d_in[11];
    const float* be2  = (const float*)d_in[12];
    const float* rm2  = (const float*)d_in[13];
    const float* rv2  = (const float*)d_in[14];
    const float* Wc   = (const float*)d_in[15];
    const float* bc   = (const float*)d_in[16];
    float* out = (float*)d_out;
    float* acc = (float*)d_ws;                 // BATCH*H1 fp32 = 256 KB

    k0_zero<<<BATCH, 256, 0, stream>>>(acc);
    k1_kron_gemm<<<NSPLIT * (H1 / NTILE), 256, 0, stream>>>(v1, v2, v3, W1, acc);
    k2_tail<<<BATCH, 128, 0, stream>>>(acc, b1, g1, be1, rm1, rv1,
                                       W2, b2, g2, be2, rm2, rv2, Wc, bc, out);
}
